// MP_GNN_32169305047060
// MI455X (gfx1250) — matrix operand order repaired and checked by reading, not compiled, not run on hardware
//
#include <hip/hip_runtime.h>

#define NN 50000
#define NE 800000
#define H  128
#define RS 136   // LDS row stride (bf16 elems): 272B rows -> 16B aligned, bank-spread

typedef __bf16 bf16_t;
typedef bf16_t v16bf __attribute__((ext_vector_type(16)));
typedef bf16_t v8bf  __attribute__((ext_vector_type(8)));
typedef float  v8f   __attribute__((ext_vector_type(8)));

enum { M_ENC_NODE = 0, M_ENC_EDGE = 1, M_UPD_EDGE = 2, M_UPD_NODE = 3, M_DECODE = 4 };

struct MLPP {
  const bf16_t *w0, *w1, *w2, *w3;      // bf16 W^T [out][Kpad]
  const float  *b0, *b1, *b2, *b3;      // fp32 biases
  const float  *g, *bb;                 // LN gamma/beta (may be null)
  int k0pad;                            // row stride of w0
  int nChunks0;                         // layer-0 K chunks of 128 (1 for K<=32)
};

struct SRCP {
  const float* x4; const float* pos; const int* send; const int* recv;
  const float* xin; const float* ein; const float* agg;
  const float* resid; float* out;
  long nrows;
};

__device__ __forceinline__ v8f vzero8() {
  v8f z;
#pragma unroll
  for (int i = 0; i < 8; ++i) z[i] = 0.f;
  return z;
}

// A fragment: 16x32 bf16. lane L(<16): row L, K = k0+0..7 (elems0-7), k0+16..23 (elems8-15)
//             lane L+16 : row L, K = k0+8..15, k0+24..31
__device__ __forceinline__ v16bf load_fragA(const bf16_t* tile, int k0) {
  const int lane = threadIdx.x & 31;
  const int r  = lane & 15;
  const int ko = k0 + ((lane & 16) ? 8 : 0);
  const bf16_t* p = tile + r * RS + ko;
  v8bf lo = *(const v8bf*)(p);
  v8bf hi = *(const v8bf*)(p + 16);
  v16bf f;
#pragma unroll
  for (int i = 0; i < 8; ++i) { f[i] = lo[i]; f[i + 8] = hi[i]; }
  return f;
}

// B fragment: 32x16 bf16 from W^T rows (row = output col). lane L(<16): col L, K = k0+0..7, k0+16..23;
//             lane L+16: col L, K = k0+8..15, k0+24..31
__device__ __forceinline__ v16bf load_fragB(const bf16_t* tile, int k0) {
  const int lane = threadIdx.x & 31;
  const int r  = lane & 15;
  const int ko = k0 + ((lane & 16) ? 8 : 0);
  const bf16_t* p = tile + r * RS + ko;
  v8bf lo = *(const v8bf*)(p);
  v8bf hi = *(const v8bf*)(p + 16);
  v16bf f;
#pragma unroll
  for (int i = 0; i < 8; ++i) { f[i] = lo[i]; f[i + 8] = hi[i]; }
  return f;
}

template <int NT>
__device__ __forceinline__ void gemm_chunk(const bf16_t* aT, const bf16_t* wT, v8f* acc, int kc) {
  for (int k0 = 0; k0 < kc; k0 += 32) {
    v16bf a = load_fragA(aT, k0);
#pragma unroll
    for (int nt = 0; nt < NT; ++nt) {
      v16bf b = load_fragB(wT + nt * 16 * RS, k0);
      acc[nt] = __builtin_amdgcn_wmma_f32_16x16x32_bf16(false, a, false, b, (short)0,
                                                        acc[nt], false, false);
    }
  }
}

// ---- staging (block-wide, 256 threads), src fp32 -> ldsA bf16 [128][RS] ----
__device__ __forceinline__ void stage128(const float* __restrict__ src, const int* __restrict__ idx,
                                         long blk0, long nrows, bf16_t* ldsA) {
  const int t = threadIdx.x;
#pragma unroll 4
  for (int it = 0; it < 32; ++it) {
    int e = it * 256 + t;
    int row = e >> 6, colp = e & 63;
    long r = blk0 + row; if (r >= nrows) r = nrows - 1;
    long sr = idx ? (long)idx[r] : r;
    const float2 v = *(const float2*)(src + sr * (long)H + colp * 2);
    bf16_t* d = ldsA + row * RS + colp * 2;
    d[0] = (bf16_t)v.x; d[1] = (bf16_t)v.y;
  }
}

__device__ __forceinline__ void stage_enc_node(const float* __restrict__ x4, long blk0, long nrows,
                                               bf16_t* ldsA) {
  const int t = threadIdx.x;
#pragma unroll
  for (int it = 0; it < 16; ++it) {
    int e = it * 256 + t;
    int row = e >> 5, col = e & 31;
    long r = blk0 + row; if (r >= nrows) r = nrows - 1;
    float v = (col < 4) ? x4[r * 4 + col] : 0.f;
    ldsA[row * RS + col] = (bf16_t)v;
  }
}

__device__ __forceinline__ void stage_enc_edge(const float* __restrict__ x4, const float* __restrict__ pos,
                                               const int* __restrict__ send, const int* __restrict__ recv,
                                               long blk0, long nrows, bf16_t* ldsA) {
  const int t = threadIdx.x;
#pragma unroll
  for (int it = 0; it < 16; ++it) {
    int e = it * 256 + t;
    int row = e >> 5, col = e & 31;
    if (col >= 8) ldsA[row * RS + col] = (bf16_t)0.f;
  }
  if (t < 128) {
    const int row = t;
    long r = blk0 + row; if (r >= nrows) r = nrows - 1;
    const int s = send[r], d = recv[r];
    float dp0 = pos[s * 3 + 0] - pos[d * 3 + 0];
    float dp1 = pos[s * 3 + 1] - pos[d * 3 + 1];
    float dp2 = pos[s * 3 + 2] - pos[d * 3 + 2];
    float nrm = sqrtf(dp0 * dp0 + dp1 * dp1 + dp2 * dp2);
    bf16_t* o = ldsA + row * RS;
    o[0] = (bf16_t)dp0; o[1] = (bf16_t)dp1; o[2] = (bf16_t)dp2; o[3] = (bf16_t)nrm;
#pragma unroll
    for (int c = 0; c < 4; ++c) o[4 + c] = (bf16_t)(x4[s * 4 + c] - x4[d * 4 + c]);
  }
}

__device__ __forceinline__ void stage_w(const bf16_t* __restrict__ wt, int srcStride, int kOff,
                                        int kc, int rows, bf16_t* ldsW) {
  const int t = threadIdx.x;
  const int totalPairs = (rows * kc) >> 1;
  for (int e = t; e < totalPairs; e += 256) {
    int row, colp;
    if (kc == 128) { row = e >> 6; colp = e & 63; }
    else           { row = e >> 4; colp = e & 15; }
    *(unsigned int*)(ldsW + row * RS + colp * 2) =
        *(const unsigned int*)(wt + (long)row * srcStride + kOff + colp * 2);
  }
}

// ---- epilogues (C layout: lane<16 -> col=lane rows 0-7; lane>=16 -> col=lane-16 rows 8-15) ----
__device__ __forceinline__ void epilogue_hidden(v8f* acc, const float* __restrict__ bias, bf16_t* aT) {
  const int lane = threadIdx.x & 31;
  const int col0 = lane & 15;
  const int rb = (lane >> 4) * 8;
#pragma unroll
  for (int nt = 0; nt < 8; ++nt) {
    const int c = nt * 16 + col0;
    const float bv = bias[c];
#pragma unroll
    for (int v = 0; v < 8; ++v) {
      float val = fmaxf(acc[nt][v] + bv, 0.f);
      aT[(rb + v) * RS + c] = (bf16_t)val;
    }
  }
}

__device__ __forceinline__ void epilogue_final(v8f* acc, const float* __restrict__ bias,
                                               const float* __restrict__ g, const float* __restrict__ bb,
                                               const float* __restrict__ resid, float* __restrict__ out,
                                               long row0, long nrows) {
  const int lane = threadIdx.x & 31;
  const int col0 = lane & 15;
  const int rb = (lane >> 4) * 8;
#pragma unroll
  for (int nt = 0; nt < 8; ++nt) {
    const float bv = bias[nt * 16 + col0];
#pragma unroll
    for (int v = 0; v < 8; ++v) acc[nt][v] += bv;
  }
#pragma unroll
  for (int v = 0; v < 8; ++v) {
    float s = 0.f, s2 = 0.f;
#pragma unroll
    for (int nt = 0; nt < 8; ++nt) { float x = acc[nt][v]; s += x; s2 += x * x; }
#pragma unroll
    for (int m = 1; m < 16; m <<= 1) {     // 16-lane butterfly (stays within half-groups)
      s  += __shfl_xor(s, m, 32);
      s2 += __shfl_xor(s2, m, 32);
    }
    const float mean = s * (1.f / 128.f);
    const float var  = s2 * (1.f / 128.f) - mean * mean;
    const float rstd = rsqrtf(var + 1e-5f);
    const long r = row0 + rb + v;
    if (r < nrows) {
#pragma unroll
      for (int nt = 0; nt < 8; ++nt) {
        const int c = nt * 16 + col0;
        float val = (acc[nt][v] - mean) * rstd * g[c] + bb[c];
        if (resid) val += resid[r * (long)H + c];
        out[r * (long)H + c] = val;
      }
    }
  }
}

// ---- fused 4-layer MLP (+ReLU, +LN, +residual) ----
template <int MODE>
__global__ __launch_bounds__(256) void mlp_fused(MLPP mp, SRCP sp) {
  __shared__ __align__(16) bf16_t ldsA[128 * RS];
  __shared__ __align__(16) bf16_t ldsW[128 * RS];
  const int wave = threadIdx.x >> 5;
  const long blk0 = (long)blockIdx.x * 128;
  const long row0 = blk0 + wave * 16;
  bf16_t* aT = ldsA + wave * 16 * RS;

  v8f acc[8];
#pragma unroll
  for (int nt = 0; nt < 8; ++nt) acc[nt] = vzero8();

  // ---- layer 0 (possibly multi-chunk K) ----
  const int kc0 = (MODE == M_ENC_NODE || MODE == M_ENC_EDGE) ? 32 : 128;
  for (int c = 0; c < mp.nChunks0; ++c) {
    __syncthreads();
    if (MODE == M_ENC_NODE)       stage_enc_node(sp.x4, blk0, sp.nrows, ldsA);
    else if (MODE == M_ENC_EDGE)  stage_enc_edge(sp.x4, sp.pos, sp.send, sp.recv, blk0, sp.nrows, ldsA);
    else if (MODE == M_UPD_EDGE) {
      if (c == 0)      stage128(sp.xin, sp.send, blk0, sp.nrows, ldsA);
      else if (c == 1) stage128(sp.xin, sp.recv, blk0, sp.nrows, ldsA);
      else             stage128(sp.ein, nullptr, blk0, sp.nrows, ldsA);
    } else if (MODE == M_UPD_NODE) stage128(c == 0 ? sp.xin : sp.agg, nullptr, blk0, sp.nrows, ldsA);
    else                           stage128(sp.xin, nullptr, blk0, sp.nrows, ldsA);
    stage_w(mp.w0, mp.k0pad, c * 128, kc0, 128, ldsW);
    __syncthreads();
    gemm_chunk<8>(aT, ldsW, acc, kc0);
  }
  epilogue_hidden(acc, mp.b0, aT);

  // ---- hidden layers 1,2 ----
  const bf16_t* ws[2] = { mp.w1, mp.w2 };
  const float*  bs[2] = { mp.b1, mp.b2 };
  for (int l = 0; l < 2; ++l) {
#pragma unroll
    for (int nt = 0; nt < 8; ++nt) acc[nt] = vzero8();
    __syncthreads();
    stage_w(ws[l], 128, 0, 128, 128, ldsW);
    __syncthreads();
    gemm_chunk<8>(aT, ldsW, acc, 128);
    epilogue_hidden(acc, bs[l], aT);
  }

  // ---- final layer ----
#pragma unroll
  for (int nt = 0; nt < 8; ++nt) acc[nt] = vzero8();
  __syncthreads();
  stage_w(mp.w3, 128, 0, 128, (MODE == M_DECODE) ? 16 : 128, ldsW);
  __syncthreads();
  if (MODE == M_DECODE) {
    gemm_chunk<1>(aT, ldsW, acc, 128);
    const int lane = threadIdx.x & 31;
    const int col = lane & 15;
    const int rb = (lane >> 4) * 8;
    if (col < 4) {
      const float bv = mp.b3[col];
#pragma unroll
      for (int v = 0; v < 8; ++v) {
        const long r = row0 + rb + v;
        if (r < sp.nrows) sp.out[r * 4 + col] = acc[0][v] + bv;
      }
    }
  } else {
    gemm_chunk<8>(aT, ldsW, acc, 128);
    epilogue_final(acc, mp.b3, mp.g, mp.bb, sp.resid, sp.out, row0, sp.nrows);
  }
}

// ---- weight convert/transpose/pad: W[K][Oc] fp32 -> W^T[Opad][Kpad] bf16 ----
__global__ __launch_bounds__(256) void prep_w(const float* __restrict__ W, bf16_t* __restrict__ WT,
                                              int K, int Kpad, int Oc, int Opad) {
  const int i = blockIdx.x * 256 + threadIdx.x;
  if (i < Opad * Kpad) {
    const int o = i / Kpad, k = i - o * Kpad;
    const float v = (o < Oc && k < K) ? W[(long)k * Oc + o] : 0.f;
    WT[(long)o * Kpad + k] = (bf16_t)v;
  }
}

// ---- segment sum: agg[recv[e]] += e_attr[e] ----
__global__ __launch_bounds__(256) void seg_sum(const float* __restrict__ eat,
                                               const int* __restrict__ recv,
                                               float* __restrict__ agg, int E) {
  const long i = (long)blockIdx.x * 256 + threadIdx.x;
  if (i < (long)E * H) {
    const long edge = i >> 7;
    const int  c = (int)(i & 127);
    atomicAdd(agg + (long)recv[edge] * H + c, eat[i]);
  }
}

extern "C" void kernel_launch(void* const* d_in, const int* in_sizes, int n_in,
                              void* d_out, int out_size, void* d_ws, size_t ws_size,
                              hipStream_t stream) {
  (void)in_sizes; (void)n_in; (void)out_size; (void)ws_size;
  const float* x4  = (const float*)d_in[0];
  const int*   ei  = (const int*)  d_in[1];
  const float* pos = (const float*)d_in[2];
  const int* send = ei;
  const int* recv = ei + NE;

  char* ws = (char*)d_ws;
  size_t off = 0;
  auto take = [&](size_t bytes) -> void* {
    void* p = ws + off;
    off += (bytes + 255) & ~(size_t)255;
    return p;
  };

  float* e_buf = (float*)take((size_t)NE * H * 4);
  float* x_buf = (float*)take((size_t)NN * H * 4);
  float* agg   = (float*)take((size_t)NN * H * 4);

  auto prep = [&](int base, int k0, int k0pad, int oLast, int oLastPad, bool ln) -> MLPP {
    MLPP mp{};
    mp.b0 = (const float*)d_in[base + 1];
    mp.b1 = (const float*)d_in[base + 3];
    mp.b2 = (const float*)d_in[base + 5];
    mp.b3 = (const float*)d_in[base + 7];
    if (ln) { mp.g = (const float*)d_in[base + 8]; mp.bb = (const float*)d_in[base + 9]; }
    bf16_t* w0t = (bf16_t*)take((size_t)128 * k0pad * 2);
    bf16_t* w1t = (bf16_t*)take((size_t)128 * 128 * 2);
    bf16_t* w2t = (bf16_t*)take((size_t)128 * 128 * 2);
    bf16_t* w3t = (bf16_t*)take((size_t)oLastPad * 128 * 2);
    auto lp = [&](const void* W, bf16_t* WT, int K, int Kpad, int Oc, int Opad) {
      const int total = Opad * Kpad;
      prep_w<<<(total + 255) / 256, 256, 0, stream>>>((const float*)W, WT, K, Kpad, Oc, Opad);
    };
    lp(d_in[base + 0], w0t, k0, k0pad, 128, 128);
    lp(d_in[base + 2], w1t, 128, 128, 128, 128);
    lp(d_in[base + 4], w2t, 128, 128, 128, 128);
    lp(d_in[base + 6], w3t, 128, 128, oLast, oLastPad);
    mp.w0 = w0t; mp.w1 = w1t; mp.w2 = w2t; mp.w3 = w3t;
    mp.k0pad = k0pad;
    mp.nChunks0 = (k0pad == 32) ? 1 : (k0pad / 128);
    return mp;
  };

  // pytree (sorted keys): edge_encoder @3, node_decoder @13, node_encoder @21,
  // processor[i] @31+20i (edge_updater +0, node_updater +10)
  MLPP enc_e = prep(3, 8, 32, 128, 128, true);
  MLPP dec   = prep(13, 128, 128, 4, 16, false);
  MLPP enc_n = prep(21, 4, 32, 128, 128, true);
  MLPP eu[4], nu[4];
  for (int i = 0; i < 4; ++i) {
    eu[i] = prep(31 + 20 * i + 0, 384, 384, 128, 128, true);
    nu[i] = prep(31 + 20 * i + 10, 256, 256, 128, 128, true);
  }

  const int gridN = (NN + 127) / 128;
  const int gridE = (NE + 127) / 128;

  SRCP base{};
  base.x4 = x4; base.pos = pos; base.send = send; base.recv = recv;

  { SRCP sp = base; sp.out = x_buf; sp.nrows = NN;
    mlp_fused<M_ENC_NODE><<<gridN, 256, 0, stream>>>(enc_n, sp); }
  { SRCP sp = base; sp.out = e_buf; sp.nrows = NE;
    mlp_fused<M_ENC_EDGE><<<gridE, 256, 0, stream>>>(enc_e, sp); }

  for (int i = 0; i < 4; ++i) {
    { SRCP sp = base; sp.xin = x_buf; sp.ein = e_buf; sp.resid = e_buf; sp.out = e_buf; sp.nrows = NE;
      mlp_fused<M_UPD_EDGE><<<gridE, 256, 0, stream>>>(eu[i], sp); }
    hipMemsetAsync(agg, 0, (size_t)NN * H * 4, stream);
    { const long total = (long)NE * H;
      seg_sum<<<(int)((total + 255) / 256), 256, 0, stream>>>(e_buf, recv, agg, NE); }
    { SRCP sp = base; sp.xin = x_buf; sp.agg = agg; sp.resid = x_buf; sp.out = x_buf; sp.nrows = NN;
      mlp_fused<M_UPD_NODE><<<gridN, 256, 0, stream>>>(nu[i], sp); }
  }
  { SRCP sp = base; sp.xin = x_buf; sp.out = (float*)d_out; sp.nrows = NN;
    mlp_fused<M_DECODE><<<gridN, 256, 0, stream>>>(dec, sp); }
}
